// PromptGRN_39470749450713
// MI455X (gfx1250) — matrix operand order repaired and checked by reading, not compiled, not run on hardware
//
#include <hip/hip_runtime.h>
#include <hip/hip_bf16.h>
#include <cmath>

typedef __attribute__((ext_vector_type(16))) __bf16 v16bf;
typedef __attribute__((ext_vector_type(8)))  float  v8f;
typedef __attribute__((ext_vector_type(4)))  unsigned int v4u;
typedef __attribute__((ext_vector_type(8)))  int    v8i;
typedef __attribute__((ext_vector_type(4)))  int    v4i;

#define WMMA_BF16(a, b, c) \
  __builtin_amdgcn_wmma_f32_16x16x32_bf16(false, (a), false, (b), (short)0, (c), false, false)

#if __has_builtin(__builtin_amdgcn_tensor_load_to_lds) && \
    __has_builtin(__builtin_amdgcn_tensor_store_from_lds) && \
    __has_builtin(__builtin_amdgcn_s_wait_tensorcnt)
#define HAVE_TDM 1
#else
#define HAVE_TDM 0
#endif

// ---------------------------------------------------------------------------
// Tiled bf16 WMMA GEMM: C[M x N] = act(A[M x K] * op(B) + bias)
// TRANSB=1: B is [N][ldb] (row-major weight, computes A @ B^T)
// TRANSB=0: B is [K][ldb] (computes A @ B)
// EPI: 0 = none, 1 = exact GELU
// Requires M%64==0, N%64==0, K%32==0.
// ---------------------------------------------------------------------------
template<int TRANSB, int EPI>
__global__ __launch_bounds__(256)
void gemm_kernel(const float* __restrict__ A, int lda,
                 const float* __restrict__ B, int ldb,
                 const float* __restrict__ bias,
                 float* __restrict__ C, int ldc, int coloff,
                 int M, int N, int K)
{
  __shared__ __bf16 As[64][34];
  __shared__ __bf16 Bs[64][34];
  const int tid  = threadIdx.x;
  const int lane = tid & 31;
  const int wave = tid >> 5;
  const int wm = wave >> 1;      // 0..3 -> row tile
  const int wn = wave & 1;       // 0..1 -> col tile pair
  const int lh = lane >> 4;      // lane half
  const int ln = lane & 15;
  const int m0 = blockIdx.x * 64;
  const int n0 = blockIdx.y * 64;

  v8f acc0 = {};
  v8f acc1 = {};

  for (int k0 = 0; k0 < K; k0 += 32) {
    for (int i = tid; i < 64 * 32; i += 256) {
      int r = i >> 5, c = i & 31;
      As[r][c] = (__bf16)A[(size_t)(m0 + r) * lda + k0 + c];
    }
    if (TRANSB) {
      for (int i = tid; i < 64 * 32; i += 256) {
        int r = i >> 5, c = i & 31;
        Bs[r][c] = (__bf16)B[(size_t)(n0 + r) * ldb + k0 + c];
      }
    } else {
      for (int i = tid; i < 64 * 32; i += 256) {
        int n = i & 63, c = i >> 6;
        Bs[n][c] = (__bf16)B[(size_t)(k0 + c) * ldb + n0 + n];
      }
    }
    __syncthreads();

    // A fragment: row m = ln; K = 16*lh + {0..15}
    v16bf fa;
    {
      const __bf16* ar = &As[wm * 16 + ln][0];
      #pragma unroll
      for (int i = 0; i < 8; ++i) {
        fa[i]     = ar[lh * 16 + i];
        fa[8 + i] = ar[lh * 16 + 8 + i];
      }
    }
    // B fragments: col n = ln; K = 16*lh + {0..15}
    #pragma unroll
    for (int j = 0; j < 2; ++j) {
      const __bf16* br = &Bs[wn * 32 + j * 16 + ln][0];
      v16bf fb;
      #pragma unroll
      for (int c = 0; c < 16; ++c) fb[c] = br[lh * 16 + c];
      if (j == 0) acc0 = WMMA_BF16(fa, fb, acc0);
      else        acc1 = WMMA_BF16(fa, fb, acc1);
    }
    __syncthreads();
  }

  #pragma unroll
  for (int j = 0; j < 2; ++j) {
    v8f acc = j ? acc1 : acc0;
    int col = n0 + wn * 32 + j * 16 + ln;
    float bv = bias ? bias[col] : 0.f;
    #pragma unroll
    for (int v = 0; v < 8; ++v) {
      int row = m0 + wm * 16 + lh * 8 + v;
      float x = acc[v] + bv;
      if (EPI == 1) x = 0.5f * x * (1.f + erff(x * 0.70710678f));
      C[(size_t)row * ldc + coloff + col] = x;
    }
  }
}

// ---------------------------------------------------------------------------
// GAT: per-row source/dest scores  s_src[i]=h_i.a[:64], s_dst[i]=h_i.a[64:]
// ---------------------------------------------------------------------------
__global__ __launch_bounds__(128)
void gat_score_kernel(const float* __restrict__ h, const float* __restrict__ a,
                      float* __restrict__ ssrc, float* __restrict__ sdst, int N)
{
  int lane = threadIdx.x & 31;
  int row  = blockIdx.x * 4 + (threadIdx.x >> 5);
  const float* hr = h + (size_t)row * 64;
  float s0 = hr[lane] * a[lane] + hr[lane + 32] * a[lane + 32];
  float s1 = hr[lane] * a[64 + lane] + hr[lane + 32] * a[96 + lane];
  #pragma unroll
  for (int o = 16; o >= 1; o >>= 1) {
    s0 += __shfl_xor(s0, o, 32);
    s1 += __shfl_xor(s1, o, 32);
  }
  if (lane == 0) { ssrc[row] = s0; sdst[row] = s1; }
}

// ---------------------------------------------------------------------------
// GAT masked attention (online softmax), fused leaky -> L2 norm -> +b -> (elu)
// one wave per row; lane handles feature dims 2*lane, 2*lane+1
// ---------------------------------------------------------------------------
__global__ __launch_bounds__(128)
void gat_attn_kernel(const float* __restrict__ h, const float* __restrict__ ssrc,
                     const float* __restrict__ sdst, const int* __restrict__ adj,
                     const float* __restrict__ bvec, float* __restrict__ out,
                     int ldo, int coloff, int N, int elu_flag)
{
  int lane = threadIdx.x & 31;
  int row  = blockIdx.x * 4 + (threadIdx.x >> 5);
  const int d0 = lane * 2;
  float si = ssrc[row];
  const int* arow = adj + (size_t)row * N;
  float m = -INFINITY, l = 0.f, a0 = 0.f, a1 = 0.f;
  for (int j = 0; j < N; ++j) {
    if ((j & 255) == 0 && j + 256 < N) __builtin_prefetch(arow + j + 256, 0, 1);
    if (arow[j] > 0) {
      float e = si + sdst[j];
      e = e > 0.f ? e : 0.2f * e;            // leaky relu
      float mn = fmaxf(m, e);
      float sc = __expf(m - mn);
      float p  = __expf(e - mn);
      const float* hj = h + (size_t)j * 64;
      l  = l * sc + p;
      a0 = a0 * sc + p * hj[d0];
      a1 = a1 * sc + p * hj[d0 + 1];
      m = mn;
    }
  }
  float inv = 1.f / fmaxf(l, 1e-30f);
  float v0 = a0 * inv, v1 = a1 * inv;
  v0 = v0 > 0.f ? v0 : 0.2f * v0;
  v1 = v1 > 0.f ? v1 : 0.2f * v1;
  float nrm = v0 * v0 + v1 * v1;
  #pragma unroll
  for (int o = 16; o >= 1; o >>= 1) nrm += __shfl_xor(nrm, o, 32);
  nrm = fmaxf(sqrtf(nrm), 1e-12f);
  v0 = v0 / nrm + bvec[d0];
  v1 = v1 / nrm + bvec[d0 + 1];
  if (elu_flag) {
    v0 = v0 > 0.f ? v0 : expm1f(v0);
    v1 = v1 > 0.f ? v1 : expm1f(v1);
  }
  out[(size_t)row * ldo + coloff + d0]     = v0;
  out[(size_t)row * ldo + coloff + d0 + 1] = v1;
}

__global__ void mean3_kernel(const float* a, const float* b, const float* c,
                             float* o, int n)
{
  int i = blockIdx.x * blockDim.x + threadIdx.x;
  if (i < n) o[i] = (a[i] + b[i] + c[i]) * (1.f / 3.f);
}

// mediator token build: [x0, family, promoter, zeros]
__global__ void med_tokens_kernel(const float* __restrict__ x0,
                                  const float* __restrict__ fam,
                                  const float* __restrict__ prom,
                                  const int* __restrict__ gid,
                                  float* __restrict__ X, int N)
{
  int i = blockIdx.x * blockDim.x + threadIdx.x;
  if (i >= N * 128) return;
  int g = i >> 7, d = i & 127;
  int id = gid[g];
  float* base = X + (size_t)g * 4 * 128;
  base[d]       = x0[d];
  base[128 + d] = fam[(size_t)id * 128 + d];
  base[256 + d] = prom[(size_t)id * 128 + d];
  base[384 + d] = 0.f;
}

// mediator attention for token 0 only: 4 heads, dh=32, 4 keys per gene
__global__ __launch_bounds__(128)
void med_attn_kernel(const float* __restrict__ qkv, float* __restrict__ out, int N)
{
  __shared__ float pbuf[4][16];
  int lane = threadIdx.x & 31;
  int w = threadIdx.x >> 5;
  int g = blockIdx.x * 4 + w;
  const float* base = qkv + (size_t)g * 4 * 384;
  int ht = lane & 15;
  int hd = ht >> 2, t = ht & 3;
  float s = 0.f;
  for (int d = 0; d < 32; ++d)
    s += base[hd * 32 + d] * base[(size_t)t * 384 + 128 + hd * 32 + d];
  s *= 0.17677669529663689f;  // 1/sqrt(32)
  float mx = s;
  mx = fmaxf(mx, __shfl_xor(mx, 1, 32));
  mx = fmaxf(mx, __shfl_xor(mx, 2, 32));
  float p = __expf(s - mx);
  float sm = p;
  sm += __shfl_xor(sm, 1, 32);
  sm += __shfl_xor(sm, 2, 32);
  p /= sm;
  if (lane < 16) pbuf[w][ht] = p;
  __syncthreads();
  #pragma unroll
  for (int i = 0; i < 4; ++i) {
    int d = lane * 4 + i;
    int hh = d >> 5;
    float acc = 0.f;
    #pragma unroll
    for (int tt = 0; tt < 4; ++tt)
      acc += pbuf[w][hh * 4 + tt] * base[(size_t)tt * 384 + 256 + d];
    out[(size_t)g * 128 + d] = acc;
  }
}

// out = LayerNorm(x + r) * g + b   (D=128; ldr=0 broadcasts residual row)
__global__ __launch_bounds__(128)
void resln_kernel(const float* __restrict__ x, int ldx,
                  const float* __restrict__ r, int ldr,
                  const float* __restrict__ g, const float* __restrict__ b,
                  float* __restrict__ out, int N)
{
  int lane = threadIdx.x & 31;
  int row  = blockIdx.x * 4 + (threadIdx.x >> 5);
  const float* xr = x + (size_t)row * ldx;
  const float* rr = r + (size_t)row * ldr;
  float v[4];
  float s = 0.f;
  #pragma unroll
  for (int i = 0; i < 4; ++i) { int d = lane + 32 * i; v[i] = xr[d] + rr[d]; s += v[i]; }
  #pragma unroll
  for (int o = 16; o >= 1; o >>= 1) s += __shfl_xor(s, o, 32);
  float mean = s * (1.f / 128.f);
  float vs = 0.f;
  #pragma unroll
  for (int i = 0; i < 4; ++i) { float t = v[i] - mean; vs += t * t; }
  #pragma unroll
  for (int o = 16; o >= 1; o >>= 1) vs += __shfl_xor(vs, o, 32);
  float rstd = rsqrtf(vs * (1.f / 128.f) + 1e-5f);
  float* orow = out + (size_t)row * 128;
  #pragma unroll
  for (int i = 0; i < 4; ++i) {
    int d = lane + 32 * i;
    orow[d] = (v[i] - mean) * rstd * g[d] + b[d];
  }
}

__global__ void concat_kernel(const float* a, int da, const float* b, int db,
                              float* o, int N)
{
  int i = blockIdx.x * blockDim.x + threadIdx.x;
  int D = da + db;
  if (i >= N * D) return;
  int r = i / D, c = i % D;
  o[i] = (c < da) ? a[(size_t)r * da + c] : b[(size_t)r * db + (c - da)];
}

// ---------------------------------------------------------------------------
// h_final writeback via Tensor Data Mover: descriptor-driven tile copy
// global -> LDS -> global, tracked by TENSORcnt. One wave per 32-row tile.
// Falls back to a plain copy if the TDM builtins are unavailable.
// ---------------------------------------------------------------------------
__global__ __launch_bounds__(32)
void tdm_copy_kernel(const float* __restrict__ src, float* __restrict__ dst,
                     int rows_per_blk)
{
#if HAVE_TDM
  __shared__ float tile[32 * 128];
  unsigned int lds = (unsigned int)(unsigned long long)(uintptr_t)&tile[0];
  unsigned long long ga =
      (unsigned long long)(uintptr_t)(src + (size_t)blockIdx.x * rows_per_blk * 128);
  unsigned long long gb =
      (unsigned long long)(uintptr_t)(dst + (size_t)blockIdx.x * rows_per_blk * 128);

  // D# group0: [1:0]=count=1 | [63:32]=lds_addr | [120:64]=global_addr | [127:126]=type 2
  v4u g0l, g0s;
  g0l[0] = 1u;  g0l[1] = lds;
  g0l[2] = (unsigned int)ga;
  g0l[3] = (unsigned int)(ga >> 32) | (2u << 30);
  g0s[0] = 1u;  g0s[1] = lds;
  g0s[2] = (unsigned int)gb;
  g0s[3] = (unsigned int)(gb >> 32) | (2u << 30);

  // D# group1: data_size=4B, tensor 128 x 4096, tile 128 x rows_per_blk, stride0=128
  const unsigned int dim0 = 128u, dim1 = 4096u;
  const unsigned int t0 = 128u, t1 = (unsigned int)rows_per_blk;
  v8i g1;
  g1[0] = (int)(2u << 16);                            // data_size = 4 bytes
  g1[1] = (int)((dim0 & 0xFFFFu) << 16);              // tensor_dim0[15:0]
  g1[2] = (int)((dim0 >> 16) | ((dim1 & 0xFFFFu) << 16));
  g1[3] = (int)((dim1 >> 16) | (t0 << 16));           // tile_dim0
  g1[4] = (int)t1;                                    // tile_dim1 (tile_dim2 = 0)
  g1[5] = (int)dim0;                                  // tensor_dim0_stride[31:0]
  g1[6] = 0;
  g1[7] = 0;
  v4i gz4 = {};  // groups 2/3 unused (2D tensor)
  v8i gz8 = {};  // trailing descriptor group (unused, zero-filled)

  __builtin_amdgcn_tensor_load_to_lds(g0l, g1, gz4, gz4, gz8, 0);
  __builtin_amdgcn_s_wait_tensorcnt(0);
  __builtin_amdgcn_tensor_store_from_lds(g0s, g1, gz4, gz4, gz8, 0);
  __builtin_amdgcn_s_wait_tensorcnt(0);
#else
  size_t base = (size_t)blockIdx.x * rows_per_blk * 128;
  for (int i = threadIdx.x; i < rows_per_blk * 128; i += 32)
    dst[base + i] = src[base + i];
#endif
}

__global__ void gather_kernel(const float* __restrict__ h,
                              const int* __restrict__ pairs,
                              float* __restrict__ o, int P)
{
  int i = blockIdx.x * blockDim.x + threadIdx.x;
  if (i >= P * 128) return;
  int p = i >> 7, d = i & 127;
  o[i] = h[(size_t)pairs[p * 2] * 128 + d];
}

__global__ __launch_bounds__(128)
void logits_kernel(const float* __restrict__ t, const float* __restrict__ h,
                   const int* __restrict__ pairs, float* __restrict__ out, int P)
{
  int lane = threadIdx.x & 31;
  int p = blockIdx.x * 4 + (threadIdx.x >> 5);
  int j = pairs[p * 2 + 1];
  const float* tr = t + (size_t)p * 128;
  const float* hr = h + (size_t)j * 128;
  float s = 0.f;
  #pragma unroll
  for (int i = 0; i < 4; ++i) s += tr[lane + 32 * i] * hr[lane + 32 * i];
  #pragma unroll
  for (int o = 16; o >= 1; o >>= 1) s += __shfl_xor(s, o, 32);
  if (lane == 0) out[p] = s;
}

// ---------------------------------------------------------------------------
// WMMA flash attention for global layers: d=128, 8 heads, dh=16, S=4096.
// One wave per 16 query rows; iterate keys in blocks of 16.
// Scores: Q(16x16, K padded to 32) x K^T via one wmma.
// P routed through LDS into A-fragment layout; P x V via one wmma.
// ---------------------------------------------------------------------------
__global__ __launch_bounds__(128)
void flash_attn_kernel(const float* __restrict__ qkv, float* __restrict__ attout,
                       int S)
{
  __shared__ __bf16 plds[4][16][18];
  const int tid  = threadIdx.x;
  const int lane = tid & 31;
  const int w  = tid >> 5;
  const int lh = lane >> 4, ln = lane & 15;
  const int head  = blockIdx.y;
  const int qbase = blockIdx.x * 64 + w * 16;
  const int qoff  = head * 16;

  // Q fragment (A layout), pre-scaled by 1/sqrt(dh)=0.25; K dims 16..31 zero
  v16bf fq;
  {
    const float* qr = qkv + (size_t)(qbase + ln) * 384 + qoff + lh * 8;
    #pragma unroll
    for (int i = 0; i < 8; ++i) {
      fq[i]     = (__bf16)(0.25f * qr[i]);
      fq[8 + i] = (__bf16)0.f;
    }
  }

  v8f o = {};
  float mrow[8], lrow[8];
  #pragma unroll
  for (int v = 0; v < 8; ++v) { mrow[v] = -INFINITY; lrow[v] = 0.f; }

  for (int kb = 0; kb < S; kb += 16) {
    // K^T fragment (B layout): K-dim = feature d (0..15 real, 16..31 zero),
    // n = key index. Element c<8 holds d=8*lh+c; c>=8 (d=16..31) zero.
    v16bf fk;
    {
      const float* kr = qkv + (size_t)(kb + ln) * 384 + 128 + qoff + lh * 8;
      #pragma unroll
      for (int c = 0; c < 16; ++c) fk[c] = c < 8 ? (__bf16)kr[c] : (__bf16)0.f;
    }
    v8f zero = {};
    v8f s = WMMA_BF16(fq, fk, zero);

    // online softmax; row m = v + 8*lh, cols across the 16-lane half
    float pv[8];
    #pragma unroll
    for (int v = 0; v < 8; ++v) {
      float sv = s[v];
      float rm = sv;
      rm = fmaxf(rm, __shfl_xor(rm, 1, 32));
      rm = fmaxf(rm, __shfl_xor(rm, 2, 32));
      rm = fmaxf(rm, __shfl_xor(rm, 4, 32));
      rm = fmaxf(rm, __shfl_xor(rm, 8, 32));
      float mn = fmaxf(mrow[v], rm);
      float sc = __expf(mrow[v] - mn);
      float p  = __expf(sv - mn);
      float ps = p;
      ps += __shfl_xor(ps, 1, 32);
      ps += __shfl_xor(ps, 2, 32);
      ps += __shfl_xor(ps, 4, 32);
      ps += __shfl_xor(ps, 8, 32);
      lrow[v] = lrow[v] * sc + ps;
      o[v] *= sc;
      mrow[v] = mn;
      pv[v] = p;
    }

    __syncthreads();
    #pragma unroll
    for (int v = 0; v < 8; ++v) plds[w][v + lh * 8][ln] = (__bf16)pv[v];
    __syncthreads();

    // P fragment (A layout): row m = ln, K = key 8*lh + {0..7}; 16..31 zero
    v16bf fp;
    #pragma unroll
    for (int i = 0; i < 8; ++i) {
      fp[i]     = plds[w][ln][lh * 8 + i];
      fp[8 + i] = (__bf16)0.f;
    }
    // V fragment (B layout): K = key (0..15 real), n = dh dim = ln
    v16bf fv;
    #pragma unroll
    for (int c = 0; c < 16; ++c)
      fv[c] = c < 8 ? (__bf16)qkv[(size_t)(kb + lh * 8 + c) * 384 + 256 + qoff + ln]
                    : (__bf16)0.f;
    o = WMMA_BF16(fp, fv, o);
  }

  #pragma unroll
  for (int v = 0; v < 8; ++v) {
    float val = o[v] / fmaxf(lrow[v], 1e-30f);
    attout[(size_t)(qbase + v + lh * 8) * 128 + qoff + ln] = val;
  }
}

// ---------------------------------------------------------------------------
// Host side
// ---------------------------------------------------------------------------
static void launch_gemm(int transB, int epi,
                        const float* A, int lda, const float* B, int ldb,
                        const float* bias, float* C, int ldc, int coloff,
                        int M, int N, int K, hipStream_t stream)
{
  dim3 grid(M / 64, N / 64);
  if (transB == 0)
    gemm_kernel<0, 0><<<grid, 256, 0, stream>>>(A, lda, B, ldb, bias, C, ldc, coloff, M, N, K);
  else if (epi == 0)
    gemm_kernel<1, 0><<<grid, 256, 0, stream>>>(A, lda, B, ldb, bias, C, ldc, coloff, M, N, K);
  else
    gemm_kernel<1, 1><<<grid, 256, 0, stream>>>(A, lda, B, ldb, bias, C, ldc, coloff, M, N, K);
}

extern "C" void kernel_launch(void* const* d_in, const int* in_sizes, int n_in,
                              void* d_out, int out_size, void* d_ws, size_t ws_size,
                              hipStream_t stream)
{
  (void)in_sizes; (void)n_in; (void)out_size; (void)ws_size;
  const int N = 4096, PE = 16384;
  const float* x_expr = (const float*)d_in[0];
  const int*   adj    = (const int*)d_in[1];
  const int*   pairs  = (const int*)d_in[2];
  const int*   gid    = (const int*)d_in[3];
  auto P = [&](int i) { return (const float*)d_in[i]; };

  float* w = (float*)d_ws;
  float* logits   = (float*)d_out;
  float* hfin_out = (float*)d_out + PE;

  // workspace layout (floats), phase-aliased; total ~12.06M floats (~48 MB)
  constexpr size_t OFF_GATH = 0;         // 3*4096*64
  constexpr size_t OFF_GATO = 786432;    // 3*4096*64
  constexpr size_t OFF_X1   = 1572864;   // 4096*192
  constexpr size_t OFF_SRC  = 2359296;   // 3*4096
  constexpr size_t OFF_DST  = 2371584;   // 3*4096
  constexpr size_t OFF_MQKV = 0;         // 16384*384 (after GAT dead)
  constexpr size_t OFF_MFF  = 0;         // 4096*512  (after MQKV dead)
  constexpr size_t OFF_MX   = 6291456;   // 16384*128
  constexpr size_t OFF_HLOC = 8388608;   // 4096*64
  constexpr size_t OFF_MATT = 8650752;   // 4096*128
  constexpr size_t OFF_MTMP = 9175040;   // 4096*128
  constexpr size_t OFF_XLN  = 9699328;   // 4096*128
  constexpr size_t OFF_PFIN = 10223616;  // 4096*128
  constexpr size_t OFF_ZCAT = 10747904;  // 4096*192
  constexpr size_t OFF_H    = 11534336;  // 4096*128
  constexpr size_t OFF_GQKV = 0;         // 4096*384
  constexpr size_t OFF_GFF  = 1572864;   // 4096*512
  constexpr size_t OFF_GATT = OFF_MATT;
  constexpr size_t OFF_GTMP = OFF_MTMP;
  constexpr size_t OFF_PHI  = 3670016;   // 16384*128
  constexpr size_t OFF_PT   = 6291456;   // 16384*128 (MX dead)

  // ---- GAT layer 1 (3 heads) -> x1 (4096 x 192, ELU applied) ----
  for (int h = 0; h < 3; ++h) {
    const float* W = P(4 + 3 * h);
    const float* a = P(5 + 3 * h);
    const float* b = P(6 + 3 * h);
    float* gh = w + OFF_GATH + (size_t)h * N * 64;
    launch_gemm(0, 0, x_expr, 512, W, 64, nullptr, gh, 64, 0, N, 64, 512, stream);
    gat_score_kernel<<<N / 4, 128, 0, stream>>>(gh, a, w + OFF_SRC + (size_t)h * N,
                                                w + OFF_DST + (size_t)h * N, N);
    gat_attn_kernel<<<N / 4, 128, 0, stream>>>(gh, w + OFF_SRC + (size_t)h * N,
                                               w + OFF_DST + (size_t)h * N, adj, b,
                                               w + OFF_X1, 192, h * 64, N, 1);
  }

  // ---- GAT layer 2 (3 heads) -> h_local (mean) ----
  for (int h = 0; h < 3; ++h) {
    const float* W = P(13 + 3 * h);
    const float* a = P(14 + 3 * h);
    const float* b = P(15 + 3 * h);
    float* gh = w + OFF_GATH + (size_t)h * N * 64;
    float* go = w + OFF_GATO + (size_t)h * N * 64;
    launch_gemm(0, 0, w + OFF_X1, 192, W, 64, nullptr, gh, 64, 0, N, 64, 192, stream);
    gat_score_kernel<<<N / 4, 128, 0, stream>>>(gh, a, w + OFF_SRC + (size_t)h * N,
                                                w + OFF_DST + (size_t)h * N, N);
    gat_attn_kernel<<<N / 4, 128, 0, stream>>>(gh, w + OFF_SRC + (size_t)h * N,
                                               w + OFF_DST + (size_t)h * N, adj, b,
                                               go, 64, 0, N, 0);
  }
  mean3_kernel<<<(N * 64 + 255) / 256, 256, 0, stream>>>(
      w + OFF_GATO, w + OFF_GATO + (size_t)N * 64, w + OFF_GATO + (size_t)2 * N * 64,
      w + OFF_HLOC, N * 64);

  // ---- Mediator transformer (token 0 only needed downstream) ----
  med_tokens_kernel<<<(N * 128 + 255) / 256, 256, 0, stream>>>(
      P(24), P(22), P(23), gid, w + OFF_MX, N);
  launch_gemm(1, 0, w + OFF_MX, 128, P(25), 128, P(26), w + OFF_MQKV, 384, 0,
              4 * N, 384, 128, stream);
  med_attn_kernel<<<N / 4, 128, 0, stream>>>(w + OFF_MQKV, w + OFF_MATT, N);
  launch_gemm(1, 0, w + OFF_MATT, 128, P(27), 128, P(28), w + OFF_MTMP, 128, 0,
              N, 128, 128, stream);
  resln_kernel<<<N / 4, 128, 0, stream>>>(w + OFF_MTMP, 128, P(24), 0,
                                          P(33), P(34), w + OFF_XLN, N);
  launch_gemm(1, 1, w + OFF_XLN, 128, P(29), 128, P(30), w + OFF_MFF, 512, 0,
              N, 512, 128, stream);
  launch_gemm(1, 0, w + OFF_MFF, 512, P(31), 512, P(32), w + OFF_MTMP, 128, 0,
              N, 128, 512, stream);
  resln_kernel<<<N / 4, 128, 0, stream>>>(w + OFF_MTMP, 128, w + OFF_XLN, 128,
                                          P(35), P(36), w + OFF_PFIN, N);

  // ---- global projection ----
  concat_kernel<<<(N * 192 + 255) / 256, 256, 0, stream>>>(
      w + OFF_HLOC, 64, w + OFF_PFIN, 128, w + OFF_ZCAT, N);
  launch_gemm(1, 0, w + OFF_ZCAT, 192, P(37), 192, P(38), w + OFF_H, 128, 0,
              N, 128, 192, stream);

  // ---- 2 global transformer layers (d=128, 8 heads, dh=16) ----
  for (int L = 0; L < 2; ++L) {
    int base = 39 + 12 * L;
    launch_gemm(1, 0, w + OFF_H, 128, P(base + 0), 128, P(base + 1),
                w + OFF_GQKV, 384, 0, N, 384, 128, stream);
    flash_attn_kernel<<<dim3(N / 64, 8), 128, 0, stream>>>(w + OFF_GQKV,
                                                           w + OFF_GATT, N);
    launch_gemm(1, 0, w + OFF_GATT, 128, P(base + 2), 128, P(base + 3),
                w + OFF_GTMP, 128, 0, N, 128, 128, stream);
    resln_kernel<<<N / 4, 128, 0, stream>>>(w + OFF_GTMP, 128, w + OFF_H, 128,
                                            P(base + 8), P(base + 9), w + OFF_H, N);
    launch_gemm(1, 1, w + OFF_H, 128, P(base + 4), 128, P(base + 5),
                w + OFF_GFF, 512, 0, N, 512, 128, stream);
    launch_gemm(1, 0, w + OFF_GFF, 512, P(base + 6), 512, P(base + 7),
                w + OFF_GTMP, 128, 0, N, 128, 512, stream);
    resln_kernel<<<N / 4, 128, 0, stream>>>(w + OFF_GTMP, 128, w + OFF_H, 128,
                                            P(base + 10), P(base + 11), w + OFF_H, N);
  }

  // ---- outputs: h_final writeback (TDM) + pair logits ----
  tdm_copy_kernel<<<N / 32, 32, 0, stream>>>(w + OFF_H, hfin_out, 32);
  gather_kernel<<<(PE * 128 + 255) / 256, 256, 0, stream>>>(w + OFF_H, pairs,
                                                            w + OFF_PHI, PE);
  launch_gemm(0, 0, w + OFF_PHI, 128, P(63), 128, nullptr, w + OFF_PT, 128, 0,
              PE, 128, 128, stream);
  logits_kernel<<<PE / 4, 128, 0, stream>>>(w + OFF_PT, w + OFF_H, pairs, logits, PE);
}
